// SenderAgent_60335700574551
// MI455X (gfx1250) — compile-verified, matrix operand order checked by reading, not run on hardware
//
#include <hip/hip_runtime.h>
#include <hip/hip_bf16.h>
#include <math.h>

// Problem sizes (fixed by the reference)
#define BB   4096   // batch
#define LL   1024   // latent / hidden
#define VV   128    // vocab
#define TT   32     // timesteps
#define KT   (LL/32) // 32 k-tiles of 32 along the reduction dim

typedef __bf16 bf16_t;
typedef __attribute__((ext_vector_type(16))) __bf16 v16bf;
typedef __attribute__((ext_vector_type(8)))  __bf16 v8bf;
typedef __attribute__((ext_vector_type(8)))  float  v8f;

// ---------------------------------------------------------------------------
// CDNA5 async copy: global -> LDS, 8 bytes per lane, tracked by ASYNCcnt.
// ---------------------------------------------------------------------------
__device__ __forceinline__ void async_copy_b64(unsigned lds_off, const void* gaddr) {
    asm volatile("global_load_async_to_lds_b64 %0, %1, off"
                 :: "v"(lds_off), "v"(gaddr) : "memory");
}
__device__ __forceinline__ void wait_asynccnt0() {
    asm volatile("s_wait_asynccnt 0x0" ::: "memory");
}

// ---------------------------------------------------------------------------
// Phase-0 kernels: pack weights into WMMA B-fragment layout (bf16), cast h0.
// B-fragment (16-bit, 32x16 KxN) per-lane layout for v_wmma_*_16x16x32_bf16:
//   lane = n_local + 16*hi (hi = lane>>4)
//   element j: k = j + 8*hi (j<8);  k = j + 8 + 8*hi (j>=8)
// Stored as pack[((ntile*KT + ktile)*32 + lane)*16 + j] -> one 32B load/lane.
// ---------------------------------------------------------------------------
__global__ __launch_bounds__(256) void pack_w_frag(const float* __restrict__ W,
                                                   bf16_t* __restrict__ P,
                                                   int nrows, int K) {
    int id = blockIdx.x * blockDim.x + threadIdx.x;
    int nTiles = nrows >> 4, kTiles = K >> 5;
    int total = nTiles * kTiles * 32;
    if (id >= total) return;
    int lane = id & 31;
    int kt   = (id >> 5) % kTiles;
    int nt   = (id >> 5) / kTiles;
    int n  = nt * 16 + (lane & 15);
    int hi = lane >> 4;
    bf16_t* dst = P + (size_t)id * 16;
    const float* src = W + (size_t)n * K + kt * 32;
#pragma unroll
    for (int j = 0; j < 16; ++j) {
        int kk = (j < 8) ? (j + 8 * hi) : (j + 8 + 8 * hi);
        dst[j] = (bf16_t)src[kk];
    }
}

__global__ __launch_bounds__(256) void f32_to_bf16_kernel(const float* __restrict__ x,
                                                          bf16_t* __restrict__ y, int n) {
    int id = blockIdx.x * blockDim.x + threadIdx.x;
    if (id < n) y[id] = (bf16_t)x[id];
}

// A-fragment from a row-major [rows][32] bf16 tile (LDS or global):
//   lane: m = lane&15, hi = lane>>4; two contiguous 16B chunks per lane.
__device__ __forceinline__ v16bf frag_from_tile(const bf16_t* tile_row, int hi) {
    v8bf lo  = *(const v8bf*)(tile_row + 8 * hi);
    v8bf hi8 = *(const v8bf*)(tile_row + 16 + 8 * hi);
    v16bf a;
#pragma unroll
    for (int j = 0; j < 8; ++j) { a[j] = lo[j]; a[j + 8] = hi8[j]; }
    return a;
}

__device__ __forceinline__ float sigf(float x) { return 1.0f / (1.0f + __expf(-x)); }

// ---------------------------------------------------------------------------
// Kernel A: gates = h_in @ W_hh^T (+ b_ih + b_hh + W_ih[:, idx_prev]) -> cell
// update -> c (fp32, in place), h_out (bf16, ping-pong).
//
// Block = 32 batch rows x 64 l-columns x all 4 gates; 8 waves:
//   wave w: qp = w&1 (gate pair {0,1} or {2,3}), wl = w>>1 (l-tile of 16).
// Each wave: 2 m-subtiles x 2 gates = 4 WMMA accumulators; each B fragment
// feeds 2 WMMAs. The 32x32 h tile is async-staged into LDS (double-buffered,
// prefetch k+1 during k) and shared by all 8 waves. k-loop is kept rolled
// and pointer-incremented so the accumulate chain coalesces in place (no
// register-rotation copies / hazard NOPs). Gate pre-activations are
// exchanged through LDS so the cell update runs on 8 contiguous-in-l
// elements per thread (coalesced c/h traffic, one msg gather per thread).
// grid = (B/32, L/64) = (128, 16)
// ---------------------------------------------------------------------------
__global__ __launch_bounds__(256) void lstm_step_kernel(
    const bf16_t* __restrict__ h_in, bf16_t* __restrict__ h_out,
    float* __restrict__ c, const bf16_t* __restrict__ whh_pack,
    const float* __restrict__ W_ih, const float* __restrict__ b_ih,
    const float* __restrict__ b_hh, const int* __restrict__ msg, int t) {
    __shared__ bf16_t atile[2][32 * 32];    // 2 x 2KB double-buffered h tile
    __shared__ float  gtile[4 * 32 * 64];   // 32KB gate pre-activations

    int tid  = threadIdx.x;
    int lane = tid & 31;
    int wave = tid >> 5;
    int qp   = wave & 1;                    // gate pair
    int wl   = wave >> 1;                   // l-tile 0..3
    int m0 = blockIdx.x * 32;
    int l0 = blockIdx.y * 64 + wl * 16;

    // async-prefetch mapping: 256 threads x 8B = 2KB tile
    int prow = tid >> 3;                    // 0..31 (batch row in tile)
    int pc8  = tid & 7;                     // 0..7  (8-byte chunk = 4 bf16)
    unsigned lds_base = (unsigned)(uintptr_t)(&atile[0][0]);
    unsigned pref_off = prow * 64 + pc8 * 8;
    const bf16_t* gsrc = h_in + (size_t)(m0 + prow) * LL + pc8 * 4;

    async_copy_b64(lds_base + pref_off, gsrc);   // k-tile 0 -> buffer 0
    gsrc += 32;

    const v8f vzero = {0.f, 0.f, 0.f, 0.f, 0.f, 0.f, 0.f, 0.f};
    v8f acc00 = vzero, acc10 = vzero, acc01 = vzero, acc11 = vzero;

    int hi = lane >> 4;
    int ml = lane & 15;
    int nbase = l0 >> 4;

    // B-fragment streams: stride 32*16 = 512 elements (1KB) per k-tile
    const bf16_t* bq0 =
        whh_pack + ((size_t)((qp * 2 + 0) * (LL >> 4) + nbase) * KT * 32 + lane) * 16;
    const bf16_t* bq1 =
        whh_pack + ((size_t)((qp * 2 + 1) * (LL >> 4) + nbase) * KT * 32 + lane) * 16;

    unsigned cur_b = 0;                     // byte offset of read tile (0 / 2048)
    const char* atile_base = (const char*)&atile[0][0];

#pragma clang loop unroll(disable)
    for (int kt = 0; kt < KT; ++kt) {
        wait_asynccnt0();                   // this wave's prefetch landed
        __syncthreads();                    // all chunks landed / prior reads done
        if (kt + 1 < KT) {
            async_copy_b64(lds_base + (cur_b ^ 2048) + pref_off, gsrc);
            gsrc += 32;
        }

        const bf16_t* trow = (const bf16_t*)(atile_base + cur_b);
        v16bf a0 = frag_from_tile(trow + ml * 32, hi);
        v16bf a1 = frag_from_tile(trow + (16 + ml) * 32, hi);
        v16bf b0 = *(const v16bf*)bq0;
        v16bf b1 = *(const v16bf*)bq1;
        acc00 = __builtin_amdgcn_wmma_f32_16x16x32_bf16(false, a0, false, b0, (short)0, acc00, false, false);
        acc10 = __builtin_amdgcn_wmma_f32_16x16x32_bf16(false, a1, false, b0, (short)0, acc10, false, false);
        acc01 = __builtin_amdgcn_wmma_f32_16x16x32_bf16(false, a0, false, b1, (short)0, acc01, false, false);
        acc11 = __builtin_amdgcn_wmma_f32_16x16x32_bf16(false, a1, false, b1, (short)0, acc11, false, false);
        bq0 += 512;
        bq1 += 512;
        cur_b ^= 2048;
    }

    // Exchange pre-activations: D layout -> m_local = r + 8*hi, n = lane&15
    {
        int llb = wl * 16 + ml;
        int q0 = qp * 2, q1 = qp * 2 + 1;
#pragma unroll
        for (int r = 0; r < 8; ++r) {
            int mA = r + 8 * hi;           // m-subtile 0
            int mB = 16 + r + 8 * hi;      // m-subtile 1
            gtile[(q0 * 32 + mA) * 64 + llb] = acc00[r];
            gtile[(q0 * 32 + mB) * 64 + llb] = acc10[r];
            gtile[(q1 * 32 + mA) * 64 + llb] = acc01[r];
            gtile[(q1 * 32 + mB) * 64 + llb] = acc11[r];
        }
    }
    __syncthreads();

    // Cell update: 8 contiguous-in-l elements per thread (coalesced)
    int base = tid * 8;                     // 0..2040 over 32m x 64l
    int m    = base >> 6;
    int llb  = base & 63;
    int b    = m0 + m;
    int lbase = blockIdx.y * 64 + llb;
    int idxp = (t > 0) ? msg[(size_t)b * TT + (t - 1)] : 0;

#pragma unroll
    for (int j = 0; j < 8; ++j) {
        int l = lbase + j;
        float pre[4];
#pragma unroll
        for (int q = 0; q < 4; ++q) {
            float p = gtile[(q * 32 + m) * 64 + llb + j] + b_ih[q * LL + l] + b_hh[q * LL + l];
            if (t > 0) p += W_ih[(size_t)(q * LL + l) * VV + idxp];  // one-hot gather
            pre[q] = p;
        }
        float ig = sigf(pre[0]);
        float fg = sigf(pre[1]);
        float gg = tanhf(pre[2]);
        float og = sigf(pre[3]);
        float cold = (t > 0) ? c[(size_t)b * LL + l] : 0.f;
        float cn = fg * cold + ig * gg;
        c[(size_t)b * LL + l] = cn;
        h_out[(size_t)b * LL + l] = (bf16_t)(og * tanhf(cn));
    }
}

// ---------------------------------------------------------------------------
// Kernel B: logits = h @ W_proj^T + b_proj -> probs (softmax) and
// idx = argmax(logits + gumbel) (straight-through token == hard one-hot).
// One block = 16 batch rows x full V=128: 8 waves, one 16x16 WMMA tile each,
// logits staged in LDS, 16-lane shuffle reductions. grid = B/16 = 256.
// ---------------------------------------------------------------------------
__global__ __launch_bounds__(256) void proj_softmax_kernel(
    const bf16_t* __restrict__ h, const bf16_t* __restrict__ wpr_pack,
    const float* __restrict__ b_proj, const float* __restrict__ u,
    int* __restrict__ msg, float* __restrict__ probs, int t) {
    __shared__ float lds[16 * VV];
    int tid  = threadIdx.x;
    int lane = tid & 31;
    int wave = tid >> 5;           // vocab tile 0..7
    int m0   = blockIdx.x * 16;
    int hi   = lane >> 4;

    v8f acc = {0.f, 0.f, 0.f, 0.f, 0.f, 0.f, 0.f, 0.f};
    const bf16_t* bq = wpr_pack + ((size_t)wave * KT * 32 + lane) * 16;
    const bf16_t* arow = h + (size_t)(m0 + (lane & 15)) * LL;
    for (int kt = 0; kt < KT; ++kt) {
        v16bf a  = frag_from_tile(arow + kt * 32, hi);
        v16bf bf = *(const v16bf*)(bq + (size_t)kt * 512);
        acc = __builtin_amdgcn_wmma_f32_16x16x32_bf16(
            false, a, false, bf, (short)0, acc, false, false);
    }

    int v  = wave * 16 + (lane & 15);
    float bp = b_proj[v];
#pragma unroll
    for (int r = 0; r < 8; ++r) lds[(r + 8 * hi) * VV + v] = acc[r] + bp;
    __syncthreads();

    // 16 threads per row (contiguous lanes -> 16-lane shuffles stay in-wave)
    int row = tid >> 4, sub = tid & 15;
    int b = m0 + row;
    float lg[8];
#pragma unroll
    for (int j = 0; j < 8; ++j) lg[j] = lds[row * VV + sub * 8 + j];

    float mx = lg[0];
#pragma unroll
    for (int j = 1; j < 8; ++j) mx = fmaxf(mx, lg[j]);
#pragma unroll
    for (int m = 1; m < 16; m <<= 1) mx = fmaxf(mx, __shfl_xor(mx, m, 16));

    float e[8], s = 0.f;
#pragma unroll
    for (int j = 0; j < 8; ++j) { e[j] = __expf(lg[j] - mx); s += e[j]; }
#pragma unroll
    for (int m = 1; m < 16; m <<= 1) s += __shfl_xor(s, m, 16);
    float inv = 1.f / s;

    size_t obase = ((size_t)b * TT + t) * VV;
#pragma unroll
    for (int j = 0; j < 8; ++j) probs[obase + sub * 8 + j] = e[j] * inv;

    // Gumbel + argmax (first-max tie break, matching jnp.argmax)
    const float* up = u + ((size_t)t * BB + b) * VV;
    float bestv = -INFINITY;
    int   besti = 0;
#pragma unroll
    for (int j = 0; j < 8; ++j) {
        int vi = sub * 8 + j;
        float g   = -__logf(-__logf(up[vi] + 1e-20f) + 1e-20f);
        float val = lg[j] + g;   // TEMPERATURE == 1
        if (val > bestv) { bestv = val; besti = vi; }
    }
#pragma unroll
    for (int m = 1; m < 16; m <<= 1) {
        float ov = __shfl_xor(bestv, m, 16);
        int   oi = __shfl_xor(besti, m, 16);
        if (ov > bestv || (ov == bestv && oi < besti)) { bestv = ov; besti = oi; }
    }
    if (sub == 0) msg[(size_t)b * TT + t] = besti;
}

// ---------------------------------------------------------------------------
// Host launcher
// ---------------------------------------------------------------------------
extern "C" void kernel_launch(void* const* d_in, const int* in_sizes, int n_in,
                              void* d_out, int out_size, void* d_ws, size_t ws_size,
                              hipStream_t stream) {
    const float* latent = (const float*)d_in[0];  // [B, L]
    const float* u      = (const float*)d_in[1];  // [T, B, V]
    const float* W_ih   = (const float*)d_in[2];  // [4L, V]
    const float* W_hh   = (const float*)d_in[3];  // [4L, L]
    const float* b_ih   = (const float*)d_in[4];  // [4L]
    const float* b_hh   = (const float*)d_in[5];  // [4L]
    const float* W_proj = (const float*)d_in[6];  // [V, L]
    const float* b_proj = (const float*)d_in[7];  // [V]

    // Outputs: message [B,T] int32, then message_probs [B,T,V] f32, flat.
    int*   msg   = (int*)d_out;
    float* probs = (float*)d_out + (size_t)BB * TT;

    // Workspace carve-up (~40.25 MiB total)
    constexpr size_t SZ_WHH = (size_t)4 * LL * LL * 2;  // 8 MiB bf16 packed W_hh
    constexpr size_t SZ_WPR = (size_t)VV * LL * 2;      // 256 KiB bf16 packed W_proj
    constexpr size_t SZ_H   = (size_t)BB * LL * 2;      // 8 MiB bf16 h buffer
    char* w = (char*)d_ws;
    bf16_t* whh_pack = (bf16_t*)w;              w += SZ_WHH;
    bf16_t* wpr_pack = (bf16_t*)w;              w += SZ_WPR;
    bf16_t* hbuf0    = (bf16_t*)w;              w += SZ_H;
    bf16_t* hbuf1    = (bf16_t*)w;              w += SZ_H;
    float*  cbuf     = (float*)w;               // 16 MiB fp32 cell state

    // Phase 0: pack weights, cast h0
    {
        int items = (4 * LL / 16) * KT * 32;  // 262144
        pack_w_frag<<<(items + 255) / 256, 256, 0, stream>>>(W_hh, whh_pack, 4 * LL, LL);
    }
    {
        int items = (VV / 16) * KT * 32;      // 8192
        pack_w_frag<<<(items + 255) / 256, 256, 0, stream>>>(W_proj, wpr_pack, VV, LL);
    }
    {
        int n = BB * LL;
        f32_to_bf16_kernel<<<(n + 255) / 256, 256, 0, stream>>>(latent, hbuf0, n);
    }

    // Recurrence: 2 kernels per step, h ping-pongs, c updated in place,
    // message[t-1] feeds the W_ih one-hot gather of step t.
    for (int t = 0; t < TT; ++t) {
        bf16_t* hin  = (t & 1) ? hbuf1 : hbuf0;
        bf16_t* hout = (t & 1) ? hbuf0 : hbuf1;
        lstm_step_kernel<<<dim3(BB / 32, LL / 64), 256, 0, stream>>>(
            hin, hout, cbuf, whh_pack, W_ih, b_ih, b_hh, msg, t);
        proj_softmax_kernel<<<BB / 16, 256, 0, stream>>>(
            hout, wpr_pack, b_proj, u, msg, probs, t);
    }
}
